// STTransformerBlock_86887188398863
// MI455X (gfx1250) — matrix operand order repaired and checked by reading, not compiled, not run on hardware
//
#include <hip/hip_runtime.h>

#define BB   16
#define TT   240
#define JJ   22
#define DD   256
#define NHH  8
#define HDD  32
#define DFFF 1024
#define NROWS (JJ*BB*TT)   // 84480, joint-major rows: r = (j*B+b)*T + t

#define DEV static __device__ __forceinline__

typedef __bf16 bf16;
typedef bf16  v16bf __attribute__((ext_vector_type(16)));
typedef bf16  v8bf  __attribute__((ext_vector_type(8)));
typedef float v8f   __attribute__((ext_vector_type(8)));

// ---------- bf16 helpers (bit-level, round-to-nearest-even) ----------
DEV unsigned short f2bf(float f) {
  unsigned int b = __float_as_uint(f);
  unsigned int r = (b + 0x7FFFu + ((b >> 16) & 1u)) >> 16;
  return (unsigned short)r;
}
DEV float bf2f(unsigned short u) { return __uint_as_float(((unsigned int)u) << 16); }

// ---------- WMMA fragment loaders (CDNA5 16x16x32 bf16 layouts) ----------
// A (16xK tile, row-major source [M][K]): lane m=l&15; K-offsets 0..15 for l<16,
// 16..31 for l>=16 -> two 16B loads.
DEV v16bf load_a_frag(const bf16* base, int ld, int m0, int k0, int lane) {
  const bf16* p = base + (size_t)(m0 + (lane & 15)) * ld + k0 + ((lane & 16) ? 16 : 0);
  v8bf lo = *(const v8bf*)p;
  v8bf hi = *(const v8bf*)(p + 8);
  v16bf r;
#pragma unroll
  for (int i = 0; i < 8; ++i) { r[i] = lo[i]; r[i + 8] = hi[i]; }
  return r;
}
// B (Kx16 tile from B^T stored [N][K] row-major): lane n=l&15 holds 16 contiguous K
// (K 0..15 for l<16, 16..31 for l>=16) -> one 32B load.
DEV v16bf load_b_frag(const bf16* base, int ld, int n0, int k0, int lane) {
  const bf16* p = base + (size_t)(n0 + (lane & 15)) * ld + k0 + ((lane & 16) ? 16 : 0);
  return *(const v16bf*)p;
}
DEV v8f wmma_bf16(v16bf a, v16bf b, v8f c) {
  return __builtin_amdgcn_wmma_f32_16x16x32_bf16(false, a, false, b, (short)0, c, false, false);
}

// ---------- layout/convert kernels ----------
// x [B,T,J,D] f32 -> xj [J,B,T,D] bf16  and res [J,B,T,D] f32 (residual base)
__global__ void pack_x_kernel(const float* __restrict__ x,
                              unsigned short* __restrict__ xj,
                              float* __restrict__ res) {
  const size_t total = (size_t)NROWS * DD;
  for (size_t idx = (size_t)blockIdx.x * 256 + threadIdx.x; idx < total;
       idx += (size_t)gridDim.x * 256) {
    size_t d = idx & 255;
    size_t r = idx >> 8;                 // (j*B+b)*T + t
    int t  = (int)(r % TT);
    int jb = (int)(r / TT);
    int b  = jb & 15;
    int j  = jb >> 4;
    size_t src = (((size_t)(b * TT + t)) * JJ + j) * DD + d;
    float v = x[src];
    xj[idx]  = f2bf(v);
    res[idx] = v;
  }
}

__global__ void cvt_kernel(const float* __restrict__ in, unsigned short* __restrict__ out, int n) {
  for (int i = blockIdx.x * 256 + threadIdx.x; i < n; i += gridDim.x * 256) out[i] = f2bf(in[i]);
}

// per-joint [J][D(d)][D(e)] f32 -> [J][e][d] bf16 (transpose so GEMM sees W^T rows [out][in])
__global__ void cvtT_perjoint_kernel(const float* __restrict__ in, unsigned short* __restrict__ out) {
  const int total = JJ * DD * DD;
  for (int i = blockIdx.x * 256 + threadIdx.x; i < total; i += gridDim.x * 256) {
    int d = i & 255;
    int rest = i >> 8;
    int e = rest & 255;
    int j = rest >> 8;
    out[i] = f2bf(in[((size_t)j * DD + d) * DD + e]);
  }
}

// ---------- generic bf16 WMMA GEMM: Y[M,N] = A[M,K] @ W^T, W stored [N][K] ----------
// Each wave computes a 32x64 tile (2 A-frags x 4 B-frags -> 8 WMMAs per k-step;
// 21.3 FLOP/byte from L2). Optional: bias, ReLU, bf16 store, f32 accumulate.
__global__ __launch_bounds__(256) void gemm_bf16_kernel(
    const unsigned short* __restrict__ Ag,
    const unsigned short* __restrict__ Wg,
    const float* __restrict__ bias,
    unsigned short* __restrict__ Ybf,
    float* __restrict__ Yacc,
    int M, int Ncol, int K,
    int rowsPerJoint,   // >0: weight block index = row0 / rowsPerJoint
    int relu) {
  const int lane = threadIdx.x & 31;
  const int wid  = threadIdx.x >> 5;
  const int colGroups = Ncol >> 6;
  const int wlin = blockIdx.x * 8 + wid;
  const int rowBlk = wlin / colGroups;
  const int cg = wlin - rowBlk * colGroups;
  if (rowBlk * 32 >= M) return;
  const int row0 = rowBlk * 32;
  const int col0 = cg * 64;

  const bf16* A = (const bf16*)Ag;
  const bf16* W = (const bf16*)Wg;
  if (rowsPerJoint > 0) W += (size_t)(row0 / rowsPerJoint) * Ncol * K;

  v8f acc[2][4] = {};
  for (int k0 = 0; k0 < K; k0 += 32) {
    v16bf a0 = load_a_frag(A, K, row0,      k0, lane);
    v16bf a1 = load_a_frag(A, K, row0 + 16, k0, lane);
#pragma unroll
    for (int tt = 0; tt < 4; ++tt) {
      v16bf b = load_b_frag(W, K, col0 + tt * 16, k0, lane);
      acc[0][tt] = wmma_bf16(a0, b, acc[0][tt]);
      acc[1][tt] = wmma_bf16(a1, b, acc[1][tt]);
    }
  }
  const int n = lane & 15;
  const int moff = (lane & 16) ? 8 : 0;
#pragma unroll
  for (int mi = 0; mi < 2; ++mi) {
#pragma unroll
    for (int tt = 0; tt < 4; ++tt) {
      int col = col0 + tt * 16 + n;
      float bv = bias ? bias[col] : 0.f;
#pragma unroll
      for (int r = 0; r < 8; ++r) {
        int row = row0 + mi * 16 + r + moff;
        float c = acc[mi][tt][r] + bv;
        if (relu) c = fmaxf(c, 0.f);
        size_t idx = (size_t)row * Ncol + col;
        if (Ybf) Ybf[idx] = f2bf(c);
        if (Yacc) Yacc[idx] += c;
      }
    }
  }
}

// ---------- temporal attention: per (j,b,h), softmax over T=240 ----------
__global__ __launch_bounds__(256) void temporal_attn_kernel(
    const unsigned short* __restrict__ Qg,
    const unsigned short* __restrict__ Kg,
    const unsigned short* __restrict__ Vg,
    unsigned short* __restrict__ Og) {
  __shared__ __align__(32) unsigned short Qs[TT * HDD];     // [t][d]
  __shared__ __align__(32) unsigned short Ks[TT * HDD];     // [s][d]
  __shared__ __align__(32) unsigned short VTs[HDD * 256];   // V^T [d][s], s padded to 256
  __shared__ __align__(32) unsigned short Ps[8][16 * 32];   // per-wave P re-layout tile

  const int wg = blockIdx.x;          // (j*B+b)*NH + h
  const int h  = wg & 7;
  const int jb = wg >> 3;
  const int tid  = threadIdx.x;
  const int lane = tid & 31, wid = tid >> 5;
  const size_t rowBase = (size_t)jb * TT;
  const int colBase = h * HDD;

  for (int idx = tid; idx < TT * HDD; idx += 256) {
    int t = idx >> 5, d = idx & 31;
    size_t g = (rowBase + t) * DD + colBase + d;
    Qs[idx] = Qg[g];
    Ks[idx] = Kg[g];
    VTs[d * 256 + t] = Vg[g];
  }
  for (int idx = tid; idx < HDD * 16; idx += 256) {   // zero pad s = 240..255
    int d = idx >> 4, s = TT + (idx & 15);
    VTs[d * 256 + s] = 0;
  }
  __syncthreads();

  const float scale = 0.17677669529663689f;   // HD^-0.5
  const bf16* QsB  = (const bf16*)Qs;
  const bf16* KsB  = (const bf16*)Ks;
  const bf16* VTsB = (const bf16*)VTs;
  bf16* PsB = (bf16*)Ps[wid];
  const int n = lane & 15;
  const int moff = (lane & 16) ? 8 : 0;

  for (int rb = wid; rb < 15; rb += 8) {
    v16bf aq = load_a_frag(QsB, HDD, rb * 16, 0, lane);   // Q block [16,32]
    float sc[15][8];
#pragma unroll
    for (int st = 0; st < 15; ++st) {
      v16bf bk = load_b_frag(KsB, HDD, st * 16, 0, lane); // K tile^T (K rows are B^T rows)
      v8f s = {};
      s = wmma_bf16(aq, bk, s);
#pragma unroll
      for (int r = 0; r < 8; ++r) sc[st][r] = s[r] * scale;
    }
    // softmax along s (row = r+moff, cols spread over 16 lanes of each half)
    float inv[8];
#pragma unroll
    for (int r = 0; r < 8; ++r) {
      float m = -1e30f;
#pragma unroll
      for (int st = 0; st < 15; ++st) m = fmaxf(m, sc[st][r]);
#pragma unroll
      for (int mm = 1; mm < 16; mm <<= 1) m = fmaxf(m, __shfl_xor(m, mm, 32));
      float s = 0.f;
#pragma unroll
      for (int st = 0; st < 15; ++st) { float e = __expf(sc[st][r] - m); sc[st][r] = e; s += e; }
#pragma unroll
      for (int mm = 1; mm < 16; mm <<= 1) s += __shfl_xor(s, mm, 32);
      inv[r] = 1.f / s;
    }
    // P @ V, bouncing P (C-layout) through LDS to A-layout, K=256 (padded)
    v8f o0 = {}, o1 = {};
#pragma unroll
    for (int u = 0; u < 8; ++u) {
#pragma unroll
      for (int hf = 0; hf < 2; ++hf) {
        int st = 2 * u + hf;
#pragma unroll
        for (int r = 0; r < 8; ++r) {
          float p = 0.f;
          if (st < 15) p = sc[st][r] * inv[r];
          ((unsigned short*)PsB)[(r + moff) * 32 + hf * 16 + n] = f2bf(p);
        }
      }
      __asm__ volatile("s_wait_dscnt 0x0" ::: "memory");   // cross-lane LDS visibility
      v16bf ap  = load_a_frag(PsB, 32, 0, 0, lane);
      v16bf bv0 = load_b_frag(VTsB, 256, 0,  u * 32, lane);
      v16bf bv1 = load_b_frag(VTsB, 256, 16, u * 32, lane);
      o0 = wmma_bf16(ap, bv0, o0);
      o1 = wmma_bf16(ap, bv1, o1);
    }
#pragma unroll
    for (int r = 0; r < 8; ++r) {
      size_t row = rowBase + rb * 16 + r + moff;
      Og[row * DD + colBase + n]      = f2bf(o0[r]);
      Og[row * DD + colBase + 16 + n] = f2bf(o1[r]);
    }
  }
}

// ---------- spatial attention: per (b,t), J=22, skeleton mask, VALU ----------
__device__ const int EDGE0[21] = {0,3,6,9,12,0,1,4,7,0,2,5,8,9,13,16,18,9,14,17,19};
__device__ const int EDGE1[21] = {3,6,9,12,15,1,4,7,10,2,5,8,11,13,16,18,20,14,17,19,21};

__global__ __launch_bounds__(256) void spatial_attn_kernel(
    const unsigned short* __restrict__ Qg,
    const unsigned short* __restrict__ Kg,
    const unsigned short* __restrict__ Vg,
    unsigned short* __restrict__ Og) {
  __shared__ unsigned short qs[JJ * DD];
  __shared__ unsigned short ks[JJ * DD];
  __shared__ unsigned short vs[JJ * DD];
  __shared__ float sc[NHH * JJ * JJ];
  __shared__ float msk[JJ * JJ];

  const int bt = blockIdx.x;
  const int b = bt / TT, t = bt - b * TT;
  const int tid = threadIdx.x;

  for (int i = tid; i < JJ * JJ; i += 256) msk[i] = -1e30f;
  __syncthreads();
  if (tid < 21) {
    msk[EDGE0[tid] * JJ + EDGE1[tid]] = 0.f;
    msk[EDGE1[tid] * JJ + EDGE0[tid]] = 0.f;
  } else if (tid >= 32 && tid < 32 + JJ) {
    int i = tid - 32;
    msk[i * JJ + i] = 0.f;
  }
  for (int idx = tid; idx < JJ * DD; idx += 256) {
    int j = idx >> 8, d = idx & 255;
    size_t r = ((size_t)(j * BB + b) * TT + t) * DD + d;
    qs[idx] = Qg[r]; ks[idx] = Kg[r]; vs[idx] = Vg[r];
  }
  __syncthreads();

  const float scale = 0.17677669529663689f;
  for (int idx = tid; idx < NHH * JJ * JJ; idx += 256) {
    int h = idx / (JJ * JJ);
    int rem = idx - h * JJ * JJ;
    int i = rem / JJ, j = rem - i * JJ;
    const unsigned short* qi = qs + i * DD + h * HDD;
    const unsigned short* kj = ks + j * DD + h * HDD;
    float s = 0.f;
#pragma unroll
    for (int d = 0; d < HDD; ++d) s += bf2f(qi[d]) * bf2f(kj[d]);
    sc[idx] = s * scale + msk[i * JJ + j];
  }
  __syncthreads();
  if (tid < NHH * JJ) {
    float* row = sc + tid * JJ;
    float m = -1e30f;
    for (int j = 0; j < JJ; ++j) m = fmaxf(m, row[j]);
    float s = 0.f;
    for (int j = 0; j < JJ; ++j) { float e = __expf(row[j] - m); row[j] = e; s += e; }
    float inv = 1.f / s;
    for (int j = 0; j < JJ; ++j) row[j] *= inv;
  }
  __syncthreads();
  for (int idx = tid; idx < NHH * JJ * HDD; idx += 256) {
    int h = idx / (JJ * HDD);
    int rem = idx - h * JJ * HDD;
    int i = rem / HDD, d = rem - i * HDD;
    const float* p = sc + (h * JJ + i) * JJ;
    float o = 0.f;
    for (int j = 0; j < JJ; ++j) o += p[j] * bf2f(vs[j * DD + h * HDD + d]);
    Og[((size_t)(i * BB + b) * TT + t) * DD + h * HDD + d] = f2bf(o);
  }
}

// ---------- layernorm: wave-per-row; optional f32/bf16 outputs, optional BTJ-permuted out ----------
__global__ __launch_bounds__(256) void layernorm_kernel(
    const float* __restrict__ X, const float* __restrict__ g, const float* __restrict__ bta,
    float* __restrict__ Yf, unsigned short* __restrict__ Ybf,
    float* __restrict__ Yperm, int M) {
  const int lane = threadIdx.x & 31;
  const int wid  = threadIdx.x >> 5;
  const int row  = blockIdx.x * 8 + wid;
  if (row >= M) return;
  const float* x = X + (size_t)row * DD;
  float v[8];
  float s = 0.f;
#pragma unroll
  for (int i = 0; i < 8; ++i) { v[i] = x[lane + i * 32]; s += v[i]; }
#pragma unroll
  for (int m = 1; m < 32; m <<= 1) s += __shfl_xor(s, m, 32);
  float mu = s * (1.f / 256.f);
  float q = 0.f;
#pragma unroll
  for (int i = 0; i < 8; ++i) { float d = v[i] - mu; q += d * d; }
#pragma unroll
  for (int m = 1; m < 32; m <<= 1) q += __shfl_xor(q, m, 32);
  float rstd = rsqrtf(q * (1.f / 256.f) + 1e-5f);
  size_t obase = 0;
  if (Yperm) {
    int j = row / (BB * TT);
    int rem = row - j * (BB * TT);       // b*T + t
    obase = ((size_t)rem * JJ + j) * DD; // (b*T+t)*J + j
  }
#pragma unroll
  for (int i = 0; i < 8; ++i) {
    int d = lane + i * 32;
    float y = (v[i] - mu) * rstd * g[d] + bta[d];
    if (Yf)    Yf[(size_t)row * DD + d] = y;
    if (Ybf)   Ybf[(size_t)row * DD + d] = f2bf(y);
    if (Yperm) Yperm[obase + d] = y;
  }
}

// ---------- host orchestration ----------
extern "C" void kernel_launch(void* const* d_in, const int* in_sizes, int n_in,
                              void* d_out, int out_size, void* d_ws, size_t ws_size,
                              hipStream_t stream) {
  (void)in_sizes; (void)n_in; (void)out_size; (void)ws_size;
  const float* x           = (const float*)d_in[0];
  const float* query_projs = (const float*)d_in[1];
  const float* key_w  = (const float*)d_in[2];
  const float* key_b  = (const float*)d_in[3];
  const float* val_w  = (const float*)d_in[4];
  const float* val_b  = (const float*)d_in[5];
  const float* sp_out_w = (const float*)d_in[6];
  const float* sp_out_b = (const float*)d_in[7];
  const float* jq = (const float*)d_in[8];
  const float* jk = (const float*)d_in[9];
  const float* jv = (const float*)d_in[10];
  const float* tp_out_w = (const float*)d_in[11];
  const float* tp_out_b = (const float*)d_in[12];
  const float* ln1_g = (const float*)d_in[13];
  const float* ln1_b = (const float*)d_in[14];
  const float* mlp_w1 = (const float*)d_in[15];
  const float* mlp_b1 = (const float*)d_in[16];
  const float* mlp_w2 = (const float*)d_in[17];
  const float* mlp_b2 = (const float*)d_in[18];
  const float* ln2_g = (const float*)d_in[19];
  const float* ln2_b = (const float*)d_in[20];

  char* ws = (char*)d_ws;
  size_t off = 0;
  auto carve = [&](size_t bytes) -> char* {
    char* p = ws + off;
    off += (bytes + 255) & ~(size_t)255;
    return p;
  };
  const size_t Nrow = (size_t)NROWS;
  unsigned short* xj  = (unsigned short*)carve(Nrow * DD * 2);
  float*          res = (float*)carve(Nrow * DD * 4);
  unsigned short* qb  = (unsigned short*)carve(Nrow * DD * 2);
  unsigned short* kb  = (unsigned short*)carve(Nrow * DD * 2);
  unsigned short* vb  = (unsigned short*)carve(Nrow * DD * 2);
  unsigned short* tb  = (unsigned short*)carve(Nrow * DD * 2);   // tout, then sout
  float*          x1f = (float*)carve(Nrow * DD * 4);
  unsigned short* x1b = (unsigned short*)carve(Nrow * DD * 2);
  unsigned short* hb  = (unsigned short*)carve(Nrow * DFFF * 2);
  unsigned short* jqT = (unsigned short*)carve((size_t)JJ * DD * DD * 2);
  unsigned short* jkT = (unsigned short*)carve((size_t)JJ * DD * DD * 2);
  unsigned short* jvT = (unsigned short*)carve((size_t)JJ * DD * DD * 2);
  unsigned short* qpT = (unsigned short*)carve((size_t)JJ * DD * DD * 2);
  unsigned short* kwB  = (unsigned short*)carve((size_t)DD * DD * 2);
  unsigned short* vwB  = (unsigned short*)carve((size_t)DD * DD * 2);
  unsigned short* spwB = (unsigned short*)carve((size_t)DD * DD * 2);
  unsigned short* tpwB = (unsigned short*)carve((size_t)DD * DD * 2);
  unsigned short* w1B  = (unsigned short*)carve((size_t)DFFF * DD * 2);
  unsigned short* w2B  = (unsigned short*)carve((size_t)DD * DFFF * 2);

  dim3 blk(256);
  pack_x_kernel<<<4096, blk, 0, stream>>>(x, xj, res);
  cvtT_perjoint_kernel<<<2048, blk, 0, stream>>>(jq, jqT);
  cvtT_perjoint_kernel<<<2048, blk, 0, stream>>>(jk, jkT);
  cvtT_perjoint_kernel<<<2048, blk, 0, stream>>>(jv, jvT);
  cvtT_perjoint_kernel<<<2048, blk, 0, stream>>>(query_projs, qpT);
  cvt_kernel<<<256, blk, 0, stream>>>(key_w, kwB, DD * DD);
  cvt_kernel<<<256, blk, 0, stream>>>(val_w, vwB, DD * DD);
  cvt_kernel<<<256, blk, 0, stream>>>(sp_out_w, spwB, DD * DD);
  cvt_kernel<<<256, blk, 0, stream>>>(tp_out_w, tpwB, DD * DD);
  cvt_kernel<<<1024, blk, 0, stream>>>(mlp_w1, w1B, DFFF * DD);
  cvt_kernel<<<1024, blk, 0, stream>>>(mlp_w2, w2B, DD * DFFF);

  const int RPJ = BB * TT;                          // rows per joint = 3840 (mult of 32)
  const int g256  = (NROWS / 32) * (DD / 64) / 8;   // 1320 blocks
  const int g1024 = (NROWS / 32) * (DFFF / 64) / 8; // 5280 blocks

  // temporal attention
  gemm_bf16_kernel<<<g256, blk, 0, stream>>>(xj, jqT, nullptr, qb, nullptr, NROWS, DD, DD, RPJ, 0);
  gemm_bf16_kernel<<<g256, blk, 0, stream>>>(xj, jkT, nullptr, kb, nullptr, NROWS, DD, DD, RPJ, 0);
  gemm_bf16_kernel<<<g256, blk, 0, stream>>>(xj, jvT, nullptr, vb, nullptr, NROWS, DD, DD, RPJ, 0);
  temporal_attn_kernel<<<JJ * BB * NHH, blk, 0, stream>>>(qb, kb, vb, tb);
  gemm_bf16_kernel<<<g256, blk, 0, stream>>>(tb, tpwB, tp_out_b, nullptr, res, NROWS, DD, DD, 0, 0);

  // spatial attention
  gemm_bf16_kernel<<<g256, blk, 0, stream>>>(xj, qpT, nullptr, qb, nullptr, NROWS, DD, DD, RPJ, 0);
  gemm_bf16_kernel<<<g256, blk, 0, stream>>>(xj, kwB, key_b, kb, nullptr, NROWS, DD, DD, 0, 0);
  gemm_bf16_kernel<<<g256, blk, 0, stream>>>(xj, vwB, val_b, vb, nullptr, NROWS, DD, DD, 0, 0);
  spatial_attn_kernel<<<BB * TT, blk, 0, stream>>>(qb, kb, vb, tb);
  gemm_bf16_kernel<<<g256, blk, 0, stream>>>(tb, spwB, sp_out_b, nullptr, res, NROWS, DD, DD, 0, 0);

  // norm1 + MLP + norm2
  layernorm_kernel<<<NROWS / 8, blk, 0, stream>>>(res, ln1_g, ln1_b, x1f, x1b, nullptr, NROWS);
  gemm_bf16_kernel<<<g1024, blk, 0, stream>>>(x1b, w1B, mlp_b1, hb, nullptr, NROWS, DFFF, DD, 0, 1);
  gemm_bf16_kernel<<<g256, blk, 0, stream>>>(hb, w2B, mlp_b2, nullptr, x1f, NROWS, DD, DFFF, 0, 0);
  layernorm_kernel<<<NROWS / 8, blk, 0, stream>>>(x1f, ln2_g, ln2_b, nullptr, nullptr,
                                                  (float*)d_out, NROWS);
}
